// WaveletVFI_20933670600851
// MI455X (gfx1250) — matrix operand order repaired and checked by reading, not compiled, not run on hardware
//
#include <hip/hip_runtime.h>
#include <hip/hip_bf16.h>
#include <cstdint>
#include <cstddef>

typedef __attribute__((ext_vector_type(16))) _Float16 v16h;
typedef __attribute__((ext_vector_type(8)))  _Float16 v8h;
typedef __attribute__((ext_vector_type(8)))  float    v8f;

#define GSTRIDE(i,n) for (long i = (long)blockIdx.x*blockDim.x + threadIdx.x; i < (n); i += (long)gridDim.x*blockDim.x)

// ---------------------------------------------------------------------------
// Implicit-GEMM conv3x3 (pad=1) with WMMA f16 -> f32 accumulate.
// D[Cout, B*OH*OW] = W[Cout, Cin*9] x im2col(X) ; fused bias/mul/lrelu/scale.
// Block: 128 threads (4 waves). Block tile: M=32, N=128.
// Each wave: 2 A-frags x 2 B-frags -> 4 WMMAs per K-step.
// A tile staged via global_load_async_to_lds_b128 (full tiles), B via im2col.
// ---------------------------------------------------------------------------
__global__ __launch_bounds__(128)
void conv_wmma_kernel(const _Float16* __restrict__ Wt, const float* __restrict__ bias,
                      const _Float16* __restrict__ X,
                      _Float16* __restrict__ out16, float* __restrict__ out32,
                      const float* __restrict__ mul,
                      int Cin, int K, int IH, int IW, int OH, int OW, int Bn,
                      int stride, int Cout, int Xoff, int XC, int Coff, int Ctot,
                      int Ntot, float scale, int act)
{
  // 112-byte row pitch: 16B-aligned ds_load_b128 fragment reads, spread banks.
  __shared__ __align__(16) _Float16 As[32][56];
  __shared__ __align__(16) _Float16 Bs[128][56];

  const int tid  = threadIdx.x;
  const int lane = tid & 31;
  const int wave = tid >> 5;
  const int m_base = blockIdx.y * 32;
  const int n_base = blockIdx.x * 128;

  // decode this thread's im2col column once
  const int gn = n_base + tid;
  const bool colok = (gn < Ntot);
  int bb = 0, oy = 0, ox = 0;
  if (colok) { bb = gn / (OH*OW); int r = gn - bb*(OH*OW); oy = r / OW; ox = r - oy*OW; }
  const int iy0 = oy*stride - 1;
  const int ix0 = ox*stride - 1;
  const _Float16* Xb = X + ((size_t)bb*XC + Xoff)*(size_t)IH*IW;

  // this thread's async A transfer: row m = tid>>2, 16B chunk c = tid&3
  const int am_ld = tid >> 2;
  const int ac_ld = (tid & 3) * 8;                 // in halfs
  const unsigned a_lds = (unsigned)(uintptr_t)&As[am_ld][ac_ld];

  v8f acc[2][2] = {{{}, {}}, {{}, {}}};

  for (int k0 = 0; k0 < K; k0 += 32) {
    __builtin_prefetch(&Wt[(size_t)m_base*K + k0 + 1024], 0, 0);

    // ---- stage A: 32x32 weight tile ----
    const bool fullA = (m_base + 32 <= Cout) && (k0 + 32 <= K);
    if (fullA) {
      // one async 16B DMA per thread: global -> LDS, overlap with B gather below
      unsigned goff = (unsigned)(((size_t)(m_base + am_ld)*K + k0 + ac_ld) * sizeof(_Float16));
      asm volatile("global_load_async_to_lds_b128 %0, %1, %2"
                   :
                   : "v"(a_lds), "v"(goff), "s"(Wt)
                   : "memory");
    } else {
      #pragma unroll
      for (int j = 0; j < 8; ++j) {
        int i  = tid + j*128;          // 0..1023
        int m  = i >> 5, kk = i & 31;
        int gm = m_base + m, gk = k0 + kk;
        _Float16 v = (_Float16)0.f;
        if (gm < Cout && gk < K) v = Wt[(size_t)gm*K + gk];
        As[m][kk] = v;
      }
    }

    // ---- stage B: 128x32 im2col tile, one N-column per thread ----
    {
      int gk = k0;
      int ci = gk / 9;                 // one divide per K-step, not per element
      int kr = gk - ci*9;
      #pragma unroll 4
      for (int kk = 0; kk < 32; ++kk, ++gk) {
        _Float16 v = (_Float16)0.f;
        if (colok && gk < K) {
          int ky = (kr >= 6) ? 2 : ((kr >= 3) ? 1 : 0);
          int kx = kr - 3*ky;
          int iy = iy0 + ky, ix = ix0 + kx;
          if ((unsigned)iy < (unsigned)IH && (unsigned)ix < (unsigned)IW)
            v = Xb[(size_t)ci*IH*IW + (size_t)iy*IW + ix];
        }
        Bs[tid][kk] = v;
        if (++kr == 9) { kr = 0; ++ci; }
      }
    }
    if (fullA) asm volatile("s_wait_asynccnt 0x0" ::: "memory");
    __syncthreads();

    // ---- A fragments (ISA 16-bit 16x32 layout), 2 M-subtiles ----
    const int am = lane & 15;
    const int ab = (lane < 16) ? 0 : 8;
    v16h afr[2];
    #pragma unroll
    for (int a = 0; a < 2; ++a) {
      v8h alo = *(const v8h*)&As[a*16 + am][ab];
      v8h ahi = *(const v8h*)&As[a*16 + am][ab + 16];
      #pragma unroll
      for (int i = 0; i < 8; ++i) { afr[a][i] = alo[i]; afr[a][i+8] = ahi[i]; }
    }

    // ---- B fragments: N-subtiles wave*32 and wave*32+16 ----
    const int bk = (lane < 16) ? 0 : 8;
    v16h bfr[2];
    #pragma unroll
    for (int t = 0; t < 2; ++t) {
      const int bn = wave*32 + t*16 + (lane & 15);
      v8h blo = *(const v8h*)&Bs[bn][bk];
      v8h bhi = *(const v8h*)&Bs[bn][bk + 16];
      #pragma unroll
      for (int i = 0; i < 8; ++i) { bfr[t][i] = blo[i]; bfr[t][i+8] = bhi[i]; }
    }

    #pragma unroll
    for (int a = 0; a < 2; ++a)
      #pragma unroll
      for (int t = 0; t < 2; ++t)
        acc[a][t] = __builtin_amdgcn_wmma_f32_16x16x32_f16(false, afr[a], false, bfr[t],
                                                           (short)0, acc[a][t], false, false);
    __syncthreads();
  }

  // ---- epilogue: D layout -> N = lane%16, M = r + 8*(lane/16) ----
  #pragma unroll
  for (int t = 0; t < 2; ++t) {
    const int ncol = n_base + wave*32 + t*16 + (lane & 15);
    if (ncol >= Ntot) continue;
    int cb = ncol / (OH*OW); int rr = ncol - cb*(OH*OW);
    const float mv = mul ? mul[ncol] : 1.f;
    #pragma unroll
    for (int a = 0; a < 2; ++a) {
      #pragma unroll
      for (int r = 0; r < 8; ++r) {
        int m = m_base + a*16 + r + 8*(lane >> 4);
        if (m >= Cout) continue;
        float y = acc[a][t][r] + bias[m];
        y *= mv;
        if (act) y = (y >= 0.f) ? y : 0.1f*y;
        y *= scale;
        size_t o = ((size_t)cb*Ctot + Coff + m)*(size_t)(OH*OW) + rr;
        if (out16) out16[o] = (_Float16)y;
        if (out32) out32[o] = y;
      }
    }
  }
}

// ---------------------------------------------------------------------------
// Elementwise kernels
// ---------------------------------------------------------------------------
__global__ void k_f32_to_f16(const float* __restrict__ in, _Float16* __restrict__ out, long n)
{
  GSTRIDE(i, n) out[i] = (_Float16)in[i];
}

// bwarp of an f16 feature tensor into a channel-offset slot of a concat buffer
__global__ void k_bwarp_f16(const _Float16* __restrict__ src, int Cs,
                            const float* __restrict__ flow,
                            _Float16* __restrict__ dst, int Coff, int Ctot,
                            int Bn, int H, int W)
{
  long n = (long)Bn*H*W;
  long hw = (long)H*W;
  GSTRIDE(i, n) {
    int x = (int)(i % W); long t = i / W; int y = (int)(t % H); int b = (int)(t / H);
    long pix = (long)y*W + x;
    float px = x + flow[(long)b*2*hw + pix];
    float py = y + flow[(long)b*2*hw + hw + pix];
    float x0f = floorf(px), y0f = floorf(py);
    float wx = px - x0f, wy = py - y0f;
    int x0 = (int)x0f, y0 = (int)y0f;
    float w00 = (1.f-wx)*(1.f-wy), w01 = wx*(1.f-wy), w10 = (1.f-wx)*wy, w11 = wx*wy;
    for (int c = 0; c < Cs; ++c) {
      const _Float16* p = src + ((long)b*Cs + c)*hw;
      float v00 = ((unsigned)y0     < (unsigned)H && (unsigned)x0     < (unsigned)W) ? (float)p[(long)y0*W + x0]       : 0.f;
      float v01 = ((unsigned)y0     < (unsigned)H && (unsigned)(x0+1) < (unsigned)W) ? (float)p[(long)y0*W + x0 + 1]   : 0.f;
      float v10 = ((unsigned)(y0+1) < (unsigned)H && (unsigned)x0     < (unsigned)W) ? (float)p[(long)(y0+1)*W + x0]   : 0.f;
      float v11 = ((unsigned)(y0+1) < (unsigned)H && (unsigned)(x0+1) < (unsigned)W) ? (float)p[(long)(y0+1)*W + x0+1] : 0.f;
      float v = v00*w00 + v01*w01 + v10*w10 + v11*w11;
      dst[((long)b*Ctot + Coff + c)*hw + pix] = (_Float16)v;
    }
  }
}

__device__ __forceinline__ void bilin3_f32(const float* img, int b, int H, int W,
                                           float px, float py, float* out3)
{
  float x0f = floorf(px), y0f = floorf(py);
  float wx = px - x0f, wy = py - y0f;
  int x0 = (int)x0f, y0 = (int)y0f;
  float w00 = (1.f-wx)*(1.f-wy), w01 = wx*(1.f-wy), w10 = (1.f-wx)*wy, w11 = wx*wy;
  long hw = (long)H*W;
  for (int c = 0; c < 3; ++c) {
    const float* p = img + ((long)b*3 + c)*hw;
    float v00 = ((unsigned)y0     < (unsigned)H && (unsigned)x0     < (unsigned)W) ? p[(long)y0*W + x0]         : 0.f;
    float v01 = ((unsigned)y0     < (unsigned)H && (unsigned)(x0+1) < (unsigned)W) ? p[(long)y0*W + x0 + 1]     : 0.f;
    float v10 = ((unsigned)(y0+1) < (unsigned)H && (unsigned)x0     < (unsigned)W) ? p[(long)(y0+1)*W + x0]     : 0.f;
    float v11 = ((unsigned)(y0+1) < (unsigned)H && (unsigned)(x0+1) < (unsigned)W) ? p[(long)(y0+1)*W + x0 + 1] : 0.f;
    out3[c] = v00*w00 + v01*w01 + v10*w10 + v11*w11;
  }
}

// builds the 8-channel f16 input of the p2 pyramid: [imgt(3), ft0/20(2), ft1/20(2), occ(1)]
__global__ void k_make_ftin(const float* __restrict__ img0, const float* __restrict__ img1,
                            const float* __restrict__ f0, const float* __restrict__ f1,
                            const float* __restrict__ occ,
                            _Float16* __restrict__ dst, int Bn, int H, int W)
{
  long n = (long)Bn*H*W;
  long hw = (long)H*W;
  GSTRIDE(i, n) {
    int x = (int)(i % W); long t = i / W; int y = (int)(t % H); int b = (int)(t / H);
    long pix = (long)y*W + x;
    float f0x = f0[(long)b*2*hw + pix],      f0y = f0[(long)b*2*hw + hw + pix];
    float f1x = f1[(long)b*2*hw + pix],      f1y = f1[(long)b*2*hw + hw + pix];
    float oc  = occ[(long)b*hw + pix];
    float a[3], c[3];
    bilin3_f32(img0, b, H, W, x + f0x, y + f0y, a);
    bilin3_f32(img1, b, H, W, x + f1x, y + f1y, c);
    _Float16* d = dst + (long)b*8*hw + pix;
    d[0*hw] = (_Float16)(oc*a[0] + (1.f-oc)*c[0]);
    d[1*hw] = (_Float16)(oc*a[1] + (1.f-oc)*c[1]);
    d[2*hw] = (_Float16)(oc*a[2] + (1.f-oc)*c[2]);
    d[3*hw] = (_Float16)(f0x * 0.05f);
    d[4*hw] = (_Float16)(f0y * 0.05f);
    d[5*hw] = (_Float16)(f1x * 0.05f);
    d[6*hw] = (_Float16)(f1y * 0.05f);
    d[7*hw] = (_Float16)oc;
  }
}

// jax.image.resize(linear, half) == 2x2 average; scale folds in the flow *0.5
__global__ void k_half_flow(const float* __restrict__ in, float* __restrict__ out,
                            int Bn, int C, int oh, int ow, float scale)
{
  long n = (long)Bn*C*oh*ow;
  GSTRIDE(i, n) {
    int x = (int)(i % ow); long t = i / ow; int y = (int)(t % oh); t /= oh;
    int c = (int)(t % C);  int b = (int)(t / C);
    int iw = 2*ow;
    const float* p = in + ((long)b*C + c)*(long)(2*oh)*iw;
    long base = (long)(2*y)*iw + 2*x;
    out[i] = 0.25f*(p[base] + p[base+1] + p[base+iw] + p[base+iw+1]) * scale;
  }
}

__global__ void k_idwt(const float* __restrict__ ll, const float* __restrict__ h9,
                       float* __restrict__ out, int Bn, int h, int w, int clip)
{
  long n = (long)Bn*3*h*w;
  long hw = (long)h*w;
  GSTRIDE(i, n) {
    int x = (int)(i % w); long t = i / w; int y = (int)(t % h); t /= h;
    int c = (int)(t % 3); int b = (int)(t / 3);
    long pix = (long)y*w + x;
    float L  = ll[i];
    const float* hp = h9 + ((long)b*9 + c*3)*hw + pix;
    float lh = hp[0], hl = hp[hw], hh = hp[2*hw];
    float tl = 0.5f*(L + lh + hl + hh);
    float tr = 0.5f*(L - lh + hl - hh);
    float bl = 0.5f*(L + lh - hl - hh);
    float br = 0.5f*(L - lh - hl + hh);
    if (clip) {
      tl = fminf(fmaxf(tl, 0.f), 1.f); tr = fminf(fmaxf(tr, 0.f), 1.f);
      bl = fminf(fmaxf(bl, 0.f), 1.f); br = fminf(fmaxf(br, 0.f), 1.f);
    }
    float* op = out + ((long)b*3 + c)*4*hw;
    long o = (long)(2*y)*(2*w) + 2*x;
    op[o] = tl; op[o+1] = tr; op[o + 2*w] = bl; op[o + 2*w + 1] = br;
  }
}

// per-(b,c) (max-min)*thresh[b] over one plane; one block per plane
__global__ void k_range(const float* __restrict__ wl, const float* __restrict__ thresh,
                        float* __restrict__ rng, int HW)
{
  __shared__ float smx[256];
  __shared__ float smn[256];
  int bc = blockIdx.x;
  const float* p = wl + (long)bc*HW;
  float mx = -3.4e38f, mn = 3.4e38f;
  for (int i = threadIdx.x; i < HW; i += blockDim.x) { float v = p[i]; mx = fmaxf(mx, v); mn = fminf(mn, v); }
  smx[threadIdx.x] = mx; smn[threadIdx.x] = mn;
  __syncthreads();
  for (int s = 128; s > 0; s >>= 1) {
    if ((int)threadIdx.x < s) {
      smx[threadIdx.x] = fmaxf(smx[threadIdx.x], smx[threadIdx.x + s]);
      smn[threadIdx.x] = fminf(smn[threadIdx.x], smn[threadIdx.x + s]);
    }
    __syncthreads();
  }
  if (threadIdx.x == 0) rng[bc] = (smx[0] - smn[0]) * thresh[bc/3];
}

// mask = up2( any_c( max_j |wh[b,c*3+j]| > rng[b,c] ) ), written at 2hp x 2wp
__global__ void k_mask_up2(const float* __restrict__ wh, const float* __restrict__ rng,
                           float* __restrict__ mask, int Bn, int hp, int wp)
{
  long n = (long)Bn*hp*wp;
  long hw = (long)hp*wp;
  GSTRIDE(i, n) {
    int x = (int)(i % wp); long t = i / wp; int y = (int)(t % hp); int b = (int)(t / hp);
    long pix = (long)y*wp + x;
    int flag = 0;
    #pragma unroll
    for (int c = 0; c < 3; ++c) {
      const float* p = wh + ((long)b*9 + c*3)*hw + pix;
      float m = fmaxf(fabsf(p[0]), fmaxf(fabsf(p[hw]), fabsf(p[2*hw])));
      flag |= (m > rng[b*3 + c]) ? 1 : 0;
    }
    float v = flag ? 1.f : 0.f;
    float* op = mask + (long)b*4*hw;
    long o = (long)(2*y)*(2*wp) + 2*x;
    op[o] = v; op[o+1] = v; op[o + 2*wp] = v; op[o + 2*wp + 1] = v;
  }
}

__global__ void k_dilate3(const float* __restrict__ in, float* __restrict__ out,
                          int Bn, int h, int w)
{
  long n = (long)Bn*h*w;
  GSTRIDE(i, n) {
    int x = (int)(i % w); long t = i / w; int y = (int)(t % h); int b = (int)(t / h);
    const float* p = in + (long)b*h*w;
    float m = -3.4e38f;
    #pragma unroll
    for (int dy = -1; dy <= 1; ++dy)
      #pragma unroll
      for (int dx = -1; dx <= 1; ++dx) {
        int yy = y + dy, xx = x + dx;
        if ((unsigned)yy < (unsigned)h && (unsigned)xx < (unsigned)w)
          m = fmaxf(m, p[(long)yy*w + xx]);
      }
    out[i] = m;
  }
}

// nearest 2x up-sample of an f16 tensor into a channel-offset slot of a concat buffer
__global__ void k_up2_f16(const _Float16* __restrict__ in, int Cs,
                          _Float16* __restrict__ dst, int Coff, int Ctot,
                          int Bn, int h, int w)
{
  long n = (long)Bn*Cs*h*w;
  GSTRIDE(i, n) {
    int x = (int)(i % w); long t = i / w; int y = (int)(t % h); t /= h;
    int c = (int)(t % Cs); int b = (int)(t / Cs);
    _Float16 v = in[i];
    _Float16* op = dst + ((long)b*Ctot + Coff + c)*(4l*h*w);
    long o = (long)(2*y)*(2*w) + 2*x;
    op[o] = v; op[o+1] = v; op[o + 2*w] = v; op[o + 2*w + 1] = v;
  }
}

// ---------------------------------------------------------------------------
// Host orchestration
// ---------------------------------------------------------------------------
static inline int cdiv(int a, int b) { return (a + b - 1) / b; }
static inline dim3 ewgrid(long n) { long b = (n + 255) / 256; if (b > 16384) b = 16384; return dim3((unsigned)b); }

struct WS { char* p; size_t off; size_t cap; };
static void* wsalloc(WS& w, size_t bytes) {
  size_t o = (w.off + 255) & ~(size_t)255;
  w.off = o + bytes;
  return w.p + o;
}

static void conv(hipStream_t st, const _Float16* Wt, const float* bias,
                 const _Float16* X, int Xoff, int XC, int Cin, int IH, int IW,
                 int Bn, int stride, int Cout,
                 _Float16* out16, float* out32, int Coff, int Ctot,
                 const float* mul, float scale, int act)
{
  int OH = (stride == 2) ? IH/2 : IH;
  int OW = (stride == 2) ? IW/2 : IW;
  int K = Cin*9, Ntot = Bn*OH*OW;
  dim3 grid(cdiv(Ntot, 128), cdiv(Cout, 32));
  conv_wmma_kernel<<<grid, 128, 0, st>>>(Wt, bias, X, out16, out32, mul,
      Cin, K, IH, IW, OH, OW, Bn, stride, Cout, Xoff, XC, Coff, Ctot, Ntot, scale, act);
}

extern "C" void kernel_launch(void* const* d_in, const int* in_sizes, int n_in,
                              void* d_out, int out_size, void* d_ws, size_t ws_size,
                              hipStream_t stream)
{
  const int Bn = 2, H0 = 256, W0 = 448;
  const int H1 = 128, W1 = 224, H2 = 64, W2 = 112, H3 = 32, W3 = 56, H4 = 16, W4 = 28;
  const long HW0 = (long)H0*W0, HW1 = (long)H1*W1, HW2 = (long)H2*W2, HW3 = (long)H3*W3, HW4 = (long)H4*W4;

  const float* img0 = (const float*)d_in[0];
  const float* img1 = (const float*)d_in[1];
  const float* ft0  = (const float*)d_in[2];
  const float* ft1  = (const float*)d_in[3];
  const float* occ  = (const float*)d_in[4];
  const float* thr  = (const float*)d_in[5];

  // params in dict order: p1_1..p1_4, p2_1..p2_4, conv4, conv4_l, conv4_h,
  //                       conv3, conv3_h, conv2, conv2_h, conv1, conv1_h  (w then b)
  static const int so[17] = {48,96,144,192, 48,96,144,192, 576,3,9, 648,9, 432,9, 216,9};
  static const int si[17] = {3,48,96,144,  8,48,96,144,  576,576,576, 1008,648, 936,432, 576,216};
  const float* pw[17]; const float* pb[17];
  for (int i = 0; i < 17; ++i) { pw[i] = (const float*)d_in[6 + 2*i]; pb[i] = (const float*)d_in[7 + 2*i]; }

  WS ws{(char*)d_ws, 0, ws_size};

  // --- pack weights to f16 [Cout][Cin*9] (OIHW is already contiguous) ---
  _Float16* wpk[17];
  for (int i = 0; i < 17; ++i) {
    long n = (long)so[i]*si[i]*9;
    wpk[i] = (_Float16*)wsalloc(ws, (size_t)n*2);
    k_f32_to_f16<<<ewgrid(n), 256, 0, stream>>>(pw[i], wpk[i], n);
  }

  // --- activation buffers (f16) ---
  _Float16* img0h = (_Float16*)wsalloc(ws, (size_t)Bn*3*HW0*2);
  _Float16* img1h = (_Float16*)wsalloc(ws, (size_t)Bn*3*HW0*2);
  _Float16* ftin  = (_Float16*)wsalloc(ws, (size_t)Bn*8*HW0*2);
  _Float16* f0_1  = (_Float16*)wsalloc(ws, (size_t)Bn*48 *HW1*2);
  _Float16* f1_1  = (_Float16*)wsalloc(ws, (size_t)Bn*48 *HW1*2);
  _Float16* f0_2  = (_Float16*)wsalloc(ws, (size_t)Bn*96 *HW2*2);
  _Float16* f1_2  = (_Float16*)wsalloc(ws, (size_t)Bn*96 *HW2*2);
  _Float16* f0_3  = (_Float16*)wsalloc(ws, (size_t)Bn*144*HW3*2);
  _Float16* f1_3  = (_Float16*)wsalloc(ws, (size_t)Bn*144*HW3*2);
  _Float16* f0_4  = (_Float16*)wsalloc(ws, (size_t)Bn*192*HW4*2);
  _Float16* f1_4  = (_Float16*)wsalloc(ws, (size_t)Bn*192*HW4*2);
  _Float16* xin4  = (_Float16*)wsalloc(ws, (size_t)Bn*576 *HW4*2);   // c4
  _Float16* xin3  = (_Float16*)wsalloc(ws, (size_t)Bn*1008*HW3*2);   // [c3 | up2(f4)]
  _Float16* xin2  = (_Float16*)wsalloc(ws, (size_t)Bn*936 *HW2*2);   // [c2 | up2(f3)]
  _Float16* xin1  = (_Float16*)wsalloc(ws, (size_t)Bn*576 *HW1*2);   // [c1 | up2(f2)]
  _Float16* f4    = (_Float16*)wsalloc(ws, (size_t)Bn*576*HW4*2);
  _Float16* f3    = (_Float16*)wsalloc(ws, (size_t)Bn*648*HW3*2);
  _Float16* f2    = (_Float16*)wsalloc(ws, (size_t)Bn*432*HW2*2);
  _Float16* f1    = (_Float16*)wsalloc(ws, (size_t)Bn*216*HW1*2);

  // --- f32 buffers ---
  float* fl0[5] = {0,0,0,0,0}; float* fl1[5] = {0,0,0,0,0};
  const long flHW[5] = {HW0, HW1, HW2, HW3, HW4};
  for (int k = 1; k <= 4; ++k) {
    fl0[k] = (float*)wsalloc(ws, (size_t)Bn*2*flHW[k]*4);
    fl1[k] = (float*)wsalloc(ws, (size_t)Bn*2*flHW[k]*4);
  }
  float* w4l = (float*)wsalloc(ws, (size_t)Bn*3*HW4*4);
  float* w4h = (float*)wsalloc(ws, (size_t)Bn*9*HW4*4);
  float* w3l = (float*)wsalloc(ws, (size_t)Bn*3*HW3*4);
  float* w3h = (float*)wsalloc(ws, (size_t)Bn*9*HW3*4);
  float* w2l = (float*)wsalloc(ws, (size_t)Bn*3*HW2*4);
  float* w2h = (float*)wsalloc(ws, (size_t)Bn*9*HW2*4);
  float* w1l = (float*)wsalloc(ws, (size_t)Bn*3*HW1*4);
  float* w1h = (float*)wsalloc(ws, (size_t)Bn*9*HW1*4);
  float* rng = (float*)wsalloc(ws, (size_t)Bn*3*4);
  float* mask3  = (float*)wsalloc(ws, (size_t)Bn*HW3*4);
  float* dmask3 = (float*)wsalloc(ws, (size_t)Bn*HW3*4);
  float* mask2  = (float*)wsalloc(ws, (size_t)Bn*HW2*4);
  float* dmask2 = (float*)wsalloc(ws, (size_t)Bn*HW2*4);
  float* mask1  = (float*)wsalloc(ws, (size_t)Bn*HW1*4);
  float* dmask1 = (float*)wsalloc(ws, (size_t)Bn*HW1*4);

  // ---- inputs to f16 ----
  k_f32_to_f16<<<ewgrid(Bn*3*HW0), 256, 0, stream>>>(img0, img0h, Bn*3*HW0);
  k_f32_to_f16<<<ewgrid(Bn*3*HW0), 256, 0, stream>>>(img1, img1h, Bn*3*HW0);
  k_make_ftin <<<ewgrid(Bn*HW0),   256, 0, stream>>>(img0, img1, ft0, ft1, occ, ftin, Bn, H0, W0);

  // ---- flow pyramids: fl[k] = resize_half(fl[k-1]) * 0.5 ----
  k_half_flow<<<ewgrid(Bn*2*HW1), 256, 0, stream>>>(ft0,    fl0[1], Bn, 2, H1, W1, 0.5f);
  k_half_flow<<<ewgrid(Bn*2*HW1), 256, 0, stream>>>(ft1,    fl1[1], Bn, 2, H1, W1, 0.5f);
  k_half_flow<<<ewgrid(Bn*2*HW2), 256, 0, stream>>>(fl0[1], fl0[2], Bn, 2, H2, W2, 0.5f);
  k_half_flow<<<ewgrid(Bn*2*HW2), 256, 0, stream>>>(fl1[1], fl1[2], Bn, 2, H2, W2, 0.5f);
  k_half_flow<<<ewgrid(Bn*2*HW3), 256, 0, stream>>>(fl0[2], fl0[3], Bn, 2, H3, W3, 0.5f);
  k_half_flow<<<ewgrid(Bn*2*HW3), 256, 0, stream>>>(fl1[2], fl1[3], Bn, 2, H3, W3, 0.5f);
  k_half_flow<<<ewgrid(Bn*2*HW4), 256, 0, stream>>>(fl0[3], fl0[4], Bn, 2, H4, W4, 0.5f);
  k_half_flow<<<ewgrid(Bn*2*HW4), 256, 0, stream>>>(fl1[3], fl1[4], Bn, 2, H4, W4, 0.5f);

  // ---- image pyramids (p1) ----
  conv(stream, wpk[0], pb[0], img0h, 0, 3,   3,   H0, W0, Bn, 2, 48,  f0_1, 0, 0, 48,  0, 1.f, 1);
  conv(stream, wpk[1], pb[1], f0_1,  0, 48,  48,  H1, W1, Bn, 2, 96,  f0_2, 0, 0, 96,  0, 1.f, 1);
  conv(stream, wpk[2], pb[2], f0_2,  0, 96,  96,  H2, W2, Bn, 2, 144, f0_3, 0, 0, 144, 0, 1.f, 1);
  conv(stream, wpk[3], pb[3], f0_3,  0, 144, 144, H3, W3, Bn, 2, 192, f0_4, 0, 0, 192, 0, 1.f, 1);
  conv(stream, wpk[0], pb[0], img1h, 0, 3,   3,   H0, W0, Bn, 2, 48,  f1_1, 0, 0, 48,  0, 1.f, 1);
  conv(stream, wpk[1], pb[1], f1_1,  0, 48,  48,  H1, W1, Bn, 2, 96,  f1_2, 0, 0, 96,  0, 1.f, 1);
  conv(stream, wpk[2], pb[2], f1_2,  0, 96,  96,  H2, W2, Bn, 2, 144, f1_3, 0, 0, 144, 0, 1.f, 1);
  conv(stream, wpk[3], pb[3], f1_3,  0, 144, 144, H3, W3, Bn, 2, 192, f1_4, 0, 0, 192, 0, 1.f, 1);

  // ---- target pyramid (p2): ft_k written straight into concat slots ----
  conv(stream, wpk[4], pb[4], ftin, 0,   8,    8,   H0, W0, Bn, 2, 48,  xin1, 0, 96,  576,  0, 1.f, 1);
  conv(stream, wpk[5], pb[5], xin1, 96,  576,  48,  H1, W1, Bn, 2, 96,  xin2, 0, 192, 936,  0, 1.f, 1);
  conv(stream, wpk[6], pb[6], xin2, 192, 936,  96,  H2, W2, Bn, 2, 144, xin3, 0, 288, 1008, 0, 1.f, 1);
  conv(stream, wpk[7], pb[7], xin3, 288, 1008, 144, H3, W3, Bn, 2, 192, xin4, 0, 384, 576,  0, 1.f, 1);

  // ---- warped features into concat slots ----
  k_bwarp_f16<<<ewgrid(Bn*HW1), 256, 0, stream>>>(f0_1, 48,  fl0[1], xin1, 0,   576,  Bn, H1, W1);
  k_bwarp_f16<<<ewgrid(Bn*HW1), 256, 0, stream>>>(f1_1, 48,  fl1[1], xin1, 48,  576,  Bn, H1, W1);
  k_bwarp_f16<<<ewgrid(Bn*HW2), 256, 0, stream>>>(f0_2, 96,  fl0[2], xin2, 0,   936,  Bn, H2, W2);
  k_bwarp_f16<<<ewgrid(Bn*HW2), 256, 0, stream>>>(f1_2, 96,  fl1[2], xin2, 96,  936,  Bn, H2, W2);
  k_bwarp_f16<<<ewgrid(Bn*HW3), 256, 0, stream>>>(f0_3, 144, fl0[3], xin3, 0,   1008, Bn, H3, W3);
  k_bwarp_f16<<<ewgrid(Bn*HW3), 256, 0, stream>>>(f1_3, 144, fl1[3], xin3, 144, 1008, Bn, H3, W3);
  k_bwarp_f16<<<ewgrid(Bn*HW4), 256, 0, stream>>>(f0_4, 192, fl0[4], xin4, 0,   576,  Bn, H4, W4);
  k_bwarp_f16<<<ewgrid(Bn*HW4), 256, 0, stream>>>(f1_4, 192, fl1[4], xin4, 192, 576,  Bn, H4, W4);

  // ---- coarsest level ----
  conv(stream, wpk[8],  pb[8],  xin4, 0, 576, 576, H4, W4, Bn, 1, 576, f4, 0,   0, 576, 0, 1.f,  1);
  conv(stream, wpk[9],  pb[9],  f4,   0, 576, 576, H4, W4, Bn, 1, 3,   0,  w4l, 0, 3,   0, 16.f, 0);
  conv(stream, wpk[10], pb[10], f4,   0, 576, 576, H4, W4, Bn, 1, 9,   0,  w4h, 0, 9,   0, 8.f,  0);
  k_idwt<<<ewgrid(Bn*3*HW4), 256, 0, stream>>>(w4l, w4h, w3l, Bn, H4, W4, 0);

  // ---- level 3 ----
  k_range   <<<Bn*3, 256, 0, stream>>>(w3l, thr, rng, (int)HW3);
  k_mask_up2<<<ewgrid(Bn*HW4), 256, 0, stream>>>(w4h, rng, mask3, Bn, H4, W4);
  k_dilate3 <<<ewgrid(Bn*HW3), 256, 0, stream>>>(mask3, dmask3, Bn, H3, W3);
  k_up2_f16 <<<ewgrid(Bn*576*HW4), 256, 0, stream>>>(f4, 576, xin3, 432, 1008, Bn, H4, W4);
  conv(stream, wpk[11], pb[11], xin3, 0, 1008, 1008, H3, W3, Bn, 1, 648, f3, 0,   0, 648, dmask3, 1.f, 1);
  conv(stream, wpk[12], pb[12], f3,   0, 648,  648,  H3, W3, Bn, 1, 9,   0,  w3h, 0, 9,   mask3,  4.f, 0);
  k_idwt<<<ewgrid(Bn*3*HW3), 256, 0, stream>>>(w3l, w3h, w2l, Bn, H3, W3, 0);

  // ---- level 2 ----
  k_range   <<<Bn*3, 256, 0, stream>>>(w2l, thr, rng, (int)HW2);
  k_mask_up2<<<ewgrid(Bn*HW3), 256, 0, stream>>>(w3h, rng, mask2, Bn, H3, W3);
  k_dilate3 <<<ewgrid(Bn*HW2), 256, 0, stream>>>(mask2, dmask2, Bn, H2, W2);
  k_up2_f16 <<<ewgrid(Bn*648*HW3), 256, 0, stream>>>(f3, 648, xin2, 288, 936, Bn, H3, W3);
  conv(stream, wpk[13], pb[13], xin2, 0, 936, 936, H2, W2, Bn, 1, 432, f2, 0,   0, 432, dmask2, 1.f, 1);
  conv(stream, wpk[14], pb[14], f2,   0, 432, 432, H2, W2, Bn, 1, 9,   0,  w2h, 0, 9,   mask2,  2.f, 0);
  k_idwt<<<ewgrid(Bn*3*HW2), 256, 0, stream>>>(w2l, w2h, w1l, Bn, H2, W2, 0);

  // ---- level 1 ----
  k_range   <<<Bn*3, 256, 0, stream>>>(w1l, thr, rng, (int)HW1);
  k_mask_up2<<<ewgrid(Bn*HW2), 256, 0, stream>>>(w2h, rng, mask1, Bn, H2, W2);
  k_dilate3 <<<ewgrid(Bn*HW1), 256, 0, stream>>>(mask1, dmask1, Bn, H1, W1);
  k_up2_f16 <<<ewgrid(Bn*432*HW2), 256, 0, stream>>>(f2, 432, xin1, 144, 576, Bn, H2, W2);
  conv(stream, wpk[15], pb[15], xin1, 0, 576, 576, H1, W1, Bn, 1, 216, f1, 0,   0, 216, dmask1, 1.f, 1);
  conv(stream, wpk[16], pb[16], f1,   0, 216, 216, H1, W1, Bn, 1, 9,   0,  w1h, 0, 9,   mask1,  1.f, 0);

  // ---- final IDWT + clip straight into d_out ----
  k_idwt<<<ewgrid(Bn*3*HW1), 256, 0, stream>>>(w1l, w1h, (float*)d_out, Bn, H1, W1, 1);
}
